// ScienceIETagger_39848706573669
// MI455X (gfx1250) — matrix operand order repaired and checked by reading, not compiled, not run on hardware
//
#include <hip/hip_runtime.h>

typedef __attribute__((ext_vector_type(16))) _Float16 v16h;
typedef __attribute__((ext_vector_type(8)))  float    v8f;

#define B_ 64
#define T_ 512
#define H_ 1024
#define K_ 8
#define M_ (B_ * T_)                     // 32768 rows
#define LOGIT_STRIDE (M_ * K_)           // 262144 floats per head
#define TAGS_OFF (3 * LOGIT_STRIDE)      // 786432
#define LOSS_OFF (TAGS_OFF + B_ * 3 * T_) // 884736
#define WPAD 1040                        // 1024 + 16 halves: keeps 32B alignment per column

// ---------------------------------------------------------------------------
// Kernel 1: fused 3-head GEMM (logits = enc @ W + b) via v_wmma_f32_16x16x32_f16
// Grid: 256 blocks x 256 threads (8 waves). Each wave computes a 16x32 tile of
// the [32768 x 24(+8 pad)] output. Weights staged f16-transposed in LDS.
// ---------------------------------------------------------------------------
__launch_bounds__(256)
__global__ void gemm_logits_kernel(const float* __restrict__ enc,
                                   const float* __restrict__ Wt, const float* __restrict__ bt,
                                   const float* __restrict__ Wp, const float* __restrict__ bpv,
                                   const float* __restrict__ Wm, const float* __restrict__ bm,
                                   float* __restrict__ out)
{
    __shared__ _Float16 lw[32 * WPAD];   // lw[n*WPAD + k], n = combined column 0..31

    const int tid = threadIdx.x;

    // Stage combined weight panel: columns 0..7 task, 8..15 process, 16..23 material, 24..31 zero
    for (int idx = tid; idx < 32 * H_; idx += 256) {
        int n = idx >> 10;
        int k = idx & (H_ - 1);
        float v = 0.0f;
        if (n < 8)       v = Wt[k * K_ + n];
        else if (n < 16) v = Wp[k * K_ + (n - 8)];
        else if (n < 24) v = Wm[k * K_ + (n - 16)];
        lw[n * WPAD + k] = (_Float16)v;
    }
    __syncthreads();

    const int lane = tid & 31;
    const int wid  = tid >> 5;
    const int m    = lane & 15;
    const int kh   = lane >> 4;               // lane half selects contiguous 16-wide K block
    const int row_base = (blockIdx.x * 8 + wid) * 16;

    const float* arow = enc + (size_t)(row_base + m) * H_;
    const _Float16* b0p = &lw[(size_t)(lane & 15) * WPAD];
    const _Float16* b1p = &lw[(size_t)((lane & 15) + 16) * WPAD];

    v8f c0 = {};
    v8f c1 = {};

    for (int k0 = 0; k0 < H_; k0 += 32) {
        // A fragment (16x32 f16): lanes 0-15 cover K = k0+[0..15];
        // lanes 16-31 cover K = k0+[16..31]   (contiguous, matches B)
        const int ka = k0 + kh * 16;
        v16h a;
#pragma unroll
        for (int e = 0; e < 8; ++e) a[e] = (_Float16)arow[ka + e];
#pragma unroll
        for (int e = 0; e < 8; ++e) a[8 + e] = (_Float16)arow[ka + 8 + e];

        // B fragments (32x16 f16): lanes 0-15 hold K = k0+[0..15], lanes 16-31 K = k0+[16..31]
        const int kb = k0 + kh * 16;
        v16h b0 = *(const v16h*)(b0p + kb);
        v16h b1 = *(const v16h*)(b1p + kb);

        c0 = __builtin_amdgcn_wmma_f32_16x16x32_f16(false, a, false, b0, (short)0, c0, false, false);
        c1 = __builtin_amdgcn_wmma_f32_16x16x32_f16(false, a, false, b1, (short)0, c1, false, false);
    }

    // Epilogue: C/D layout — VGPR r: lanes 0-15 M=r, lanes 16-31 M=r+8; N = lane&15
#pragma unroll
    for (int r = 0; r < 8; ++r) {
        const int row = row_base + r + 8 * kh;
        const int n0  = lane & 15;                 // columns 0..15: task / process
        const int kk0 = n0 & 7;
        const float bias0 = (n0 < 8) ? bt[kk0] : bpv[kk0];
        out[(size_t)(n0 >> 3) * LOGIT_STRIDE + (size_t)row * K_ + kk0] = c0[r] + bias0;

        const int n1 = (lane & 15) + 16;           // columns 16..23: material; 24..31 padding
        if (n1 < 24) {
            const int kk1 = n1 & 7;
            out[2 * (size_t)LOGIT_STRIDE + (size_t)row * K_ + kk1] = c1[r] + bm[kk1];
        }
    }

    // Zero the loss accumulator (CRF kernel runs strictly after on the stream)
    if (blockIdx.x == 0 && tid == 0) out[LOSS_OFF] = 0.0f;
}

// ---------------------------------------------------------------------------
// Kernel 2: per-(head,batch) Viterbi decode + CRF NLL. One wave per chain,
// lanes act as K=8 next-states (replicated 4x); cross-state reductions via
// wave shuffles; backpointers in LDS; loss via global_atomic_add_f32.
// Grid: 48 blocks x 128 threads (4 waves) = 192 chains.
// ---------------------------------------------------------------------------
__launch_bounds__(128)
__global__ void crf_viterbi_kernel(const int* __restrict__ labels,
                                   const int* __restrict__ len_tokens,
                                   const float* __restrict__ tr_task,
                                   const float* __restrict__ tr_proc,
                                   const float* __restrict__ tr_mat,
                                   float* __restrict__ out)
{
    __shared__ unsigned char bp_lds[4][T_ * K_];   // 4 x 4096 = 16 KB

    const int tid  = threadIdx.x;
    const int lane = tid & 31;
    const int wid  = tid >> 5;
    const int seq  = blockIdx.x * 4 + wid;          // 0..191
    const int h    = seq >> 6;                      // head 0..2
    const int b    = seq & 63;                      // batch

    const float* trans  = (h == 0) ? tr_task : (h == 1) ? tr_proc : tr_mat;
    const float* logits = out + (size_t)h * LOGIT_STRIDE + (size_t)b * T_ * K_;
    unsigned char* bptr = bp_lds[wid];

    const int j = lane & 7;
    float trc[8];                                   // trans column j: trc[i] = trans[i][j]
#pragma unroll
    for (int i = 0; i < 8; ++i) trc[i] = trans[i * 8 + j];

    // ---------------- Viterbi (all-ones mask) ----------------
    float alpha = logits[j];
    for (int t = 1; t < T_; ++t) {
        float best = -3.4e38f;
        int   arg  = 0;
#pragma unroll
        for (int i = 0; i < 8; ++i) {
            const float s = __shfl(alpha, i, 32) + trc[i];
            if (s > best) { best = s; arg = i; }    // first-max ties like jnp.argmax
        }
        if (lane < 8) bptr[t * 8 + j] = (unsigned char)arg;
        alpha = best + logits[t * 8 + j];
    }
    float afin[8];
#pragma unroll
    for (int i = 0; i < 8; ++i) afin[i] = __shfl(alpha, i, 32);
    __syncthreads();                                 // make LDS backpointers visible

    if (lane == 0) {
        int cur = 0;
        float bv = afin[0];
        for (int i = 1; i < 8; ++i) if (afin[i] > bv) { bv = afin[i]; cur = i; }
        float* tagout = out + TAGS_OFF + (size_t)b * (3 * T_) + (size_t)h * T_;
        tagout[T_ - 1] = (float)(cur + 8 * h);
        for (int t = T_ - 1; t >= 1; --t) {
            cur = bptr[t * 8 + cur];
            tagout[t - 1] = (float)(cur + 8 * h);
        }
    }

    // ---------------- CRF forward (log-Z) with length mask ----------------
    const int len = len_tokens[b];
    float la = logits[j];
    for (int t = 1; t < T_; ++t) {
        float av[8];
#pragma unroll
        for (int i = 0; i < 8; ++i) av[i] = __shfl(la, i, 32) + trc[i];
        float mx = av[0];
#pragma unroll
        for (int i = 1; i < 8; ++i) mx = fmaxf(mx, av[i]);
        float s = 0.0f;
#pragma unroll
        for (int i = 0; i < 8; ++i) s += __expf(av[i] - mx);
        const float na = mx + __logf(s) + logits[t * 8 + j];
        la = (t < len) ? na : la;
    }
    float fv[8];
#pragma unroll
    for (int i = 0; i < 8; ++i) fv[i] = __shfl(la, i, 32);
    float mx2 = fv[0];
#pragma unroll
    for (int i = 1; i < 8; ++i) mx2 = fmaxf(mx2, fv[i]);
    float s2 = 0.0f;
#pragma unroll
    for (int i = 0; i < 8; ++i) s2 += __expf(fv[i] - mx2);
    const float logZ = mx2 + __logf(s2);

    // ---------------- emit / transition scores (lane-parallel over t) ----------------
    const int* lab = labels + (size_t)b * (3 * T_) + (size_t)h * T_;
    float emit = 0.0f, tsc = 0.0f;
    for (int t = lane; t < T_; t += 32) {
        const int tag = lab[t] - 8 * h;
        if (t < len) {
            emit += logits[t * 8 + tag];
            if (t >= 1) {
                const int tp = lab[t - 1] - 8 * h;
                tsc += trans[tp * 8 + tag];
            }
        }
    }
#pragma unroll
    for (int off = 16; off > 0; off >>= 1) {
        emit += __shfl_xor(emit, off, 32);
        tsc  += __shfl_xor(tsc,  off, 32);
    }
    if (lane == 0) atomicAdd(out + LOSS_OFF, -(emit + tsc - logZ));
}

// ---------------------------------------------------------------------------
extern "C" void kernel_launch(void* const* d_in, const int* in_sizes, int n_in,
                              void* d_out, int out_size, void* d_ws, size_t ws_size,
                              hipStream_t stream)
{
    (void)in_sizes; (void)n_in; (void)out_size; (void)d_ws; (void)ws_size;

    const float* enc        = (const float*)d_in[0];
    const int*   labels     = (const int*)d_in[1];
    const int*   len_tokens = (const int*)d_in[2];
    const float* Wt         = (const float*)d_in[3];
    const float* bt         = (const float*)d_in[4];
    const float* Wp         = (const float*)d_in[5];
    const float* bpv        = (const float*)d_in[6];
    const float* Wm         = (const float*)d_in[7];
    const float* bm         = (const float*)d_in[8];
    const float* tr_task    = (const float*)d_in[9];
    const float* tr_proc    = (const float*)d_in[10];
    const float* tr_mat     = (const float*)d_in[11];
    float*       out        = (float*)d_out;

    // 32768 rows / (8 waves * 16 rows) = 256 blocks
    gemm_logits_kernel<<<256, 256, 0, stream>>>(enc, Wt, bt, Wp, bpv, Wm, bm, out);
    // 192 chains / 4 waves per block = 48 blocks
    crf_viterbi_kernel<<<48, 128, 0, stream>>>(labels, len_tokens, tr_task, tr_proc, tr_mat, out);
}
